// Mixer_27728308863797
// MI455X (gfx1250) — compile-verified, matrix operand order checked by reading, not run on hardware
//
#include <hip/hip_runtime.h>
#include <hip/hip_bf16.h>

// ---------------------------------------------------------------------------
// Mamba block for MI455X (gfx1250), round 4.
//
//  * bf16 operands pre-converted in workspace (no cvt VALU in GEMM hot loop)
//  * Double-buffered LDS + CDNA5 async global->LDS staging
//    (global_load_async_to_lds_b128 / s_wait_asynccnt).
//    Builtin signature (from clang diagnostics):
//      (int4 addrspace(1)*, int4 addrspace(3)*, imm offset, imm cpol)
//  * 128x128 block tile, wave = 32x64 strip = 8 WMMAs/K-step.
// ---------------------------------------------------------------------------

typedef __attribute__((ext_vector_type(16))) __bf16 v16bf;
typedef __attribute__((ext_vector_type(8)))  float  v8f;
typedef __attribute__((__vector_size__(16))) int    v4i_;

__device__ __forceinline__ __bf16 f2bf(float f) {
    // round-to-nearest-even f32 -> bf16, pure bit manipulation
    unsigned u = __builtin_bit_cast(unsigned, f);
    u = (u + 0x7FFFu + ((u >> 16) & 1u)) >> 16;
    unsigned short h = (unsigned short)u;
    return __builtin_bit_cast(__bf16, h);
}

// ---- async global->LDS staging path (CDNA5) -------------------------------
#if defined(__has_builtin)
#if __has_builtin(__builtin_amdgcn_global_load_async_to_lds_b128) && \
    __has_builtin(__builtin_amdgcn_s_wait_asynccnt)
#define USE_ASYNC_LDS 1
#endif
#endif

#ifdef USE_ASYNC_LDS
#define COPY16(l, g)                                                     \
    __builtin_amdgcn_global_load_async_to_lds_b128(                      \
        (__attribute__((address_space(1))) v4i_*)(void*)(g),             \
        (__attribute__((address_space(3))) v4i_*)(void*)(l), 0, 0)
#define WAIT_ASYNC(n) __builtin_amdgcn_s_wait_asynccnt(n)
#else
#define COPY16(l, g) __builtin_memcpy((l), (g), 16)
#define WAIT_ASYNC(n) ((void)0)
#endif

// ---------------------------------------------------------------------------
// Tiled WMMA GEMM:  C[M,N](f32) = A[M,K](bf16, row stride lda)
//                               @ W[N,K]^T(bf16, row stride ldw)
// Block: 128(M) x 128(N), K-step 32, 256 threads = 8 waves (4 m x 2 n).
// Wave computes a 32x64 strip: 2 A-fragments x 4 B-fragments = 8 WMMAs/step.
// Double-buffered LDS, async staging pipelined one K-slab ahead.
// M % 128 == 0 and K % 32 == 0 for all calls; N bounds-handled by clamping
// (duplicate rows feed unstored output columns only).
// ---------------------------------------------------------------------------
#define GB_M 128
#define GB_N 128
#define GB_K 32
#define LDS_STRIDE 40            // bf16 elems; 80B pitch -> conflict-free frags
#define A_TILE (GB_M * LDS_STRIDE)
#define B_TILE (GB_N * LDS_STRIDE)

__global__ void __launch_bounds__(256)
gemm_bf16_wmma(const __bf16* __restrict__ A, int lda,
               const __bf16* __restrict__ W, int ldw,
               float* __restrict__ C, int ldc,
               int N, int K)
{
    __shared__ __align__(16) __bf16 As[2 * A_TILE];
    __shared__ __align__(16) __bf16 Bs[2 * B_TILE];

    const int tid  = threadIdx.x;
    const int m0   = blockIdx.y * GB_M;
    const int n0   = blockIdx.x * GB_N;
    const int lane = tid & 31;
    const int wid  = tid >> 5;
    const int wm   = wid & 3;     // 32-row strip
    const int wn   = wid >> 2;    // 64-col strip
    const int half = lane >> 4;   // K offset +8 for upper half-wave
    const int l16  = lane & 15;

    v8f acc[2][4] = {};

    const int nk = K / GB_K;

    // stage one 128x32 A slab + 128x32 B slab into LDS buffer `buf`.
    // 512 16-byte chunks each; every thread issues exactly 2 A + 2 B copies
    // (B row clamped, never skipped -> fixed 4 async ops/wave/stage).
    auto stage = [&](int buf, int k0) {
        #pragma unroll
        for (int s = 0; s < 2; ++s) {
            const int c   = tid + s * 256;      // 0..511
            const int row = c >> 2;             // 0..127
            const int kch = (c & 3) << 3;       // 0,8,16,24
            {
                const __bf16* g = A + (size_t)(m0 + row) * lda + (k0 + kch);
                __bf16* l = &As[buf * A_TILE + row * LDS_STRIDE + kch];
                COPY16(l, g);
            }
            {
                int n = n0 + row;
                if (n > N - 1) n = N - 1;       // clamp, don't branch
                const __bf16* g = W + (size_t)n * ldw + (k0 + kch);
                __bf16* l = &Bs[buf * B_TILE + row * LDS_STRIDE + kch];
                COPY16(l, g);
            }
        }
    };

    stage(0, 0);

    for (int kb = 0; kb < nk; ++kb) {
        const int cur = kb & 1;
        if (kb + 1 < nk) {
            stage(cur ^ 1, (kb + 1) * GB_K);    // prefetch next slab
            WAIT_ASYNC(4);                      // previous slab complete
        } else {
            WAIT_ASYNC(0);
        }
        __syncthreads();

        const __bf16* Ab = &As[cur * A_TILE];
        const __bf16* Bb = &Bs[cur * B_TILE];

        // ISA 16-bit fragment layout: vector elems 0..7  -> K = 8*half + 0..7
        //                             vector elems 8..15 -> K = 8*half + 16..23
        v16bf af[2], bfr[4];
        #pragma unroll
        for (int r = 0; r < 2; ++r) {
            const __bf16* p =
                &Ab[(wm * 32 + r * 16 + l16) * LDS_STRIDE + (half << 3)];
            __builtin_memcpy(&af[r], p, 16);
            __builtin_memcpy(((char*)&af[r]) + 16, p + 16, 16);
        }
        #pragma unroll
        for (int j = 0; j < 4; ++j) {
            const __bf16* p =
                &Bb[(wn * 64 + j * 16 + l16) * LDS_STRIDE + (half << 3)];
            __builtin_memcpy(&bfr[j], p, 16);
            __builtin_memcpy(((char*)&bfr[j]) + 16, p + 16, 16);
        }
        #pragma unroll
        for (int j = 0; j < 4; ++j) {
            acc[0][j] = __builtin_amdgcn_wmma_f32_16x16x32_bf16(
                false, af[0], false, bfr[j], (short)0, acc[0][j], false, false);
            acc[1][j] = __builtin_amdgcn_wmma_f32_16x16x32_bf16(
                false, af[1], false, bfr[j], (short)0, acc[1][j], false, false);
        }
        __syncthreads();
    }

    // C layout per ISA: VGPR r -> row r (lanes 0-15) / r+8 (lanes 16-31)
    #pragma unroll
    for (int r2 = 0; r2 < 2; ++r2) {
        #pragma unroll
        for (int j = 0; j < 4; ++j) {
            const int col = n0 + wn * 64 + j * 16 + l16;
            if (col < N) {
                #pragma unroll
                for (int r = 0; r < 8; ++r) {
                    const int row = m0 + wm * 32 + r2 * 16 + r + (half << 3);
                    C[(size_t)row * ldc + col] = acc[r2][j][r];
                }
            }
        }
    }
}

// ---------------------------------------------------------------------------
// Generic strided f32 -> bf16 converter (weights / activations).
// ---------------------------------------------------------------------------
__global__ void __launch_bounds__(256)
convert_bf16_kernel(const float* __restrict__ src, int src_ld,
                    __bf16* __restrict__ dst, int dst_ld,
                    unsigned total, unsigned cols)
{
    const unsigned idx = blockIdx.x * 256u + threadIdx.x;
    if (idx >= total) return;
    const unsigned r = idx / cols;
    const unsigned c = idx - r * cols;
    dst[(size_t)r * dst_ld + c] = f2bf(src[(size_t)r * src_ld + c]);
}

// ---------------------------------------------------------------------------
// Depthwise causal conv (D_CONV=4) + bias + SiLU; writes f32 (scan/gating)
// and bf16 (x_proj GEMM operand).
// ---------------------------------------------------------------------------
__global__ void __launch_bounds__(256)
conv_silu_kernel(const float* __restrict__ xz, const float* __restrict__ conv_w,
                 const float* __restrict__ conv_b,
                 float* __restrict__ xs, __bf16* __restrict__ xs16)
{
    const size_t idx = (size_t)blockIdx.x * 256 + threadIdx.x;  // M*2048
    const int    d   = (int)(idx & 2047);
    const size_t m   = idx >> 11;
    const int    l   = (int)(m & 4095);
    const size_t mb  = m - (size_t)l;

    float sum = conv_b[d];
    #pragma unroll
    for (int k = 0; k < 4; ++k) {
        const int li = l + k - 3;
        if (li >= 0)
            sum += xz[(mb + li) * 4096 + d] * conv_w[d * 4 + k];
    }
    const float v = sum / (1.0f + __expf(-sum));   // SiLU
    xs[idx]   = v;
    xs16[idx] = f2bf(v);
}

// ---------------------------------------------------------------------------
// dt = softplus(dt_raw + dt_bias), in place.
// ---------------------------------------------------------------------------
__global__ void __launch_bounds__(256)
softplus_kernel(float* __restrict__ dt, const float* __restrict__ dt_bias)
{
    const size_t idx = (size_t)blockIdx.x * 256 + threadIdx.x;  // M*2048
    const int d = (int)(idx & 2047);
    const float v = dt[idx] + dt_bias[d];
    dt[idx] = (v > 20.0f) ? v : log1pf(__expf(v));
}

// ---------------------------------------------------------------------------
// Selective scan: per (b, d) channel, h[16] in registers, sequential over L.
// ---------------------------------------------------------------------------
__global__ void __launch_bounds__(256)
scan_kernel(const float* __restrict__ dt, const float* __restrict__ xs,
            const float* __restrict__ xdbl, const float* __restrict__ A_log,
            float* __restrict__ ys)
{
    const int d = blockIdx.x * 256 + threadIdx.x;  // 0..2047
    const int b = blockIdx.y;                      // 0..1

    float Aa[16], h[16];
    #pragma unroll
    for (int n = 0; n < 16; ++n) {
        Aa[n] = -__expf(A_log[d * 16 + n]);
        h[n]  = 0.0f;
    }

    for (int t = 0; t < 4096; ++t) {
        const size_t m   = (size_t)b * 4096 + t;
        const float  dtv = dt[m * 2048 + d];
        const float  xv  = xs[m * 2048 + d];
        const float  dtx = dtv * xv;
        const float* bc  = xdbl + m * 96;   // cols 64..79 = B_t, 80..95 = C_t
        float y = 0.0f;
        #pragma unroll
        for (int n = 0; n < 16; ++n) {
            const float dA = __expf(dtv * Aa[n]);
            h[n] = dA * h[n] + dtx * bc[64 + n];
            y += h[n] * bc[80 + n];
        }
        ys[m * 2048 + d] = y;
    }
}

// ---------------------------------------------------------------------------
// y16 = bf16( (ys + xs*D_in) * silu(z) )  -- operand for out_proj GEMM.
// ---------------------------------------------------------------------------
__global__ void __launch_bounds__(256)
combine_kernel(const float* __restrict__ ys, const float* __restrict__ xs,
               const float* __restrict__ xz, const float* __restrict__ D_in,
               __bf16* __restrict__ y16)
{
    const size_t idx = (size_t)blockIdx.x * 256 + threadIdx.x;  // M*2048
    const int    d   = (int)(idx & 2047);
    const size_t m   = idx >> 11;
    const float  z   = xz[m * 4096 + 2048 + d];
    const float  sig = 1.0f / (1.0f + __expf(-z));
    y16[idx] = f2bf((ys[idx] + xs[idx] * D_in[d]) * (z * sig));
}

// ---------------------------------------------------------------------------
// out += D_skip * u
// ---------------------------------------------------------------------------
__global__ void __launch_bounds__(256)
skip_kernel(float* __restrict__ out, const float* __restrict__ u,
            const float* __restrict__ D_skip)
{
    const size_t idx = (size_t)blockIdx.x * 256 + threadIdx.x;  // M*1024
    const int c = (int)(idx & 1023);
    out[idx] += D_skip[c] * u[idx];
}

// ---------------------------------------------------------------------------

extern "C" void kernel_launch(void* const* d_in, const int* in_sizes, int n_in,
                              void* d_out, int out_size, void* d_ws, size_t ws_size,
                              hipStream_t stream) {
    (void)in_sizes; (void)n_in; (void)out_size; (void)ws_size;

    const float* u       = (const float*)d_in[0];   // [2,4096,1024]
    const float* in_w    = (const float*)d_in[1];   // [4096,1024]
    const float* conv_w  = (const float*)d_in[2];   // [2048,4]
    const float* conv_b  = (const float*)d_in[3];   // [2048]
    const float* xp_w    = (const float*)d_in[4];   // [96,2048]
    const float* dtp_w   = (const float*)d_in[5];   // [2048,64]
    const float* dt_bias = (const float*)d_in[6];   // [2048]
    const float* A_log   = (const float*)d_in[7];   // [2048,16]
    const float* D_in_v  = (const float*)d_in[8];   // [2048]
    const float* out_w   = (const float*)d_in[9];   // [1024,2048]
    const float* D_skip  = (const float*)d_in[10];  // [1024]
    float* out = (float*)d_out;                     // [2,4096,1024]

    const int M = 8192;  // B*L

    // ---- workspace layout (256B-aligned regions) ----
    char* p = (char*)d_ws;
    auto alloc = [&](size_t bytes) {
        char* r = p;
        p += (bytes + 255) & ~(size_t)255;
        return r;
    };
    float*  xz    = (float*) alloc((size_t)M * 4096 * 4);  // x | z
    float*  xs    = (float*) alloc((size_t)M * 2048 * 4);  // conv+silu (f32)
    float*  xdbl  = (float*) alloc((size_t)M * 96   * 4);  // [dt_r | B | C]
    float*  dt    = (float*) alloc((size_t)M * 2048 * 4);
    float*  ys    = (float*) alloc((size_t)M * 2048 * 4);  // scan out
    __bf16* u16   = (__bf16*)alloc((size_t)M * 1024 * 2);
    __bf16* wi16  = (__bf16*)alloc((size_t)4096 * 1024 * 2);
    __bf16* xs16  = (__bf16*)alloc((size_t)M * 2048 * 2);
    __bf16* wxp16 = (__bf16*)alloc((size_t)96 * 2048 * 2);
    __bf16* dtr16 = (__bf16*)alloc((size_t)M * 64 * 2);
    __bf16* wdt16 = (__bf16*)alloc((size_t)2048 * 64 * 2);
    __bf16* y16   = (__bf16*)alloc((size_t)M * 2048 * 2);
    __bf16* wo16  = (__bf16*)alloc((size_t)1024 * 2048 * 2);

    const dim3 blk(256);
    auto cvt = [&](const float* s, int sld, __bf16* dp, int dld,
                   unsigned rows, unsigned cols) {
        const unsigned total = rows * cols;
        convert_bf16_kernel<<<dim3((total + 255) / 256), blk, 0, stream>>>(
            s, sld, dp, dld, total, cols);
    };

    // 0) bf16 operand copies
    cvt(u,    1024, u16,  1024, 8192, 1024);
    cvt(in_w, 1024, wi16, 1024, 4096, 1024);

    // 1) xz = u @ in_proj_w^T          (N=4096, K=1024)
    gemm_bf16_wmma<<<dim3(4096 / GB_N, M / GB_M), blk, 0, stream>>>(
        u16, 1024, wi16, 1024, xz, 4096, 4096, 1024);

    // 2) x = silu(conv(x) + b)  -> f32 + bf16
    conv_silu_kernel<<<dim3((unsigned)((size_t)M * 2048 / 256)), blk, 0, stream>>>(
        xz, conv_w, conv_b, xs, xs16);

    // 3) x_dbl = x @ x_proj_w^T        (N=96, K=2048)
    cvt(xp_w, 2048, wxp16, 2048, 96, 2048);
    gemm_bf16_wmma<<<dim3(1, M / GB_M), blk, 0, stream>>>(
        xs16, 2048, wxp16, 2048, xdbl, 96, 96, 2048);

    // 4) dt_raw = x_dbl[:, :64] @ dt_proj_w^T   (N=2048, K=64)
    cvt(xdbl, 96, dtr16, 64, 8192, 64);          // strided slice -> bf16
    cvt(dtp_w, 64, wdt16, 64, 2048, 64);
    gemm_bf16_wmma<<<dim3(2048 / GB_N, M / GB_M), blk, 0, stream>>>(
        dtr16, 64, wdt16, 64, dt, 2048, 2048, 64);

    // 5) dt = softplus(dt_raw + dt_bias)
    softplus_kernel<<<dim3((unsigned)((size_t)M * 2048 / 256)), blk, 0, stream>>>(
        dt, dt_bias);

    // 6) selective scan over L
    scan_kernel<<<dim3(2048 / 256, 2), blk, 0, stream>>>(dt, xs, xdbl, A_log, ys);

    // 7) y = (ys + x*D_in) * silu(z)  -> bf16
    combine_kernel<<<dim3((unsigned)((size_t)M * 2048 / 256)), blk, 0, stream>>>(
        ys, xs, xz, D_in_v, y16);

    // 8) out = y @ out_proj_w^T        (N=1024, K=2048)
    cvt(out_w, 2048, wo16, 2048, 1024, 2048);
    gemm_bf16_wmma<<<dim3(1024 / GB_N, M / GB_M), blk, 0, stream>>>(
        y16, 2048, wo16, 2048, out, 1024, 1024, 2048);

    // 9) out += D_skip * u
    skip_kernel<<<dim3((unsigned)((size_t)M * 1024 / 256)), blk, 0, stream>>>(
        out, u, D_skip);
}
